// Transformer_54004918780714
// MI455X (gfx1250) — compile-verified, matrix operand order checked by reading, not run on hardware
//
#include <hip/hip_runtime.h>
#include <hip/hip_bf16.h>
#include <math.h>

#define LYR 6
#define DM  512
#define NH  8
#define HDIM 64
#define FFD 2048
#define BBATCH 4
#define SSEQ 384
#define TSEQ 384
#define MTOK (BBATCH * SSEQ)   /* 1536 token rows (same for src and tgt) */
#define NVOCAB 32000

typedef __attribute__((ext_vector_type(16))) _Float16 v16h;
typedef __attribute__((ext_vector_type(8)))  _Float16 v8h;
typedef __attribute__((ext_vector_type(8)))  float    v8f;
typedef __attribute__((ext_vector_type(4)))  float    v4f;

// ---------------------------------------------------------------------------
// WMMA helpers (CDNA5 wave32, 16x16x32 f16 -> f32)
// A fragment: 16x32 f16. lane l (l&15 = row), k-half = (l>>4)*8.
//   elements 0..7  -> K = kb .. kb+7
//   elements 8..15 -> K = kb+16 .. kb+23      (kb in {0,8})
// B^T fragment uses the identical pattern with rows = N columns.
// C/D: lane l, vgpr v -> M = v + 8*(l>>4), N = l&15.
// ---------------------------------------------------------------------------
__device__ __forceinline__ v16h load_frag(const _Float16* p, int ld) {
  const int lane = threadIdx.x & 31;
  const _Float16* row = p + (size_t)(lane & 15) * ld + ((lane >> 4) << 3);
  v8h lo = *(const v8h*)(row);
  v8h hi = *(const v8h*)(row + 16);
  v16h f;
#pragma unroll
  for (int i = 0; i < 8; ++i) { f[i] = lo[i]; f[i + 8] = hi[i]; }
  return f;
}

__device__ __forceinline__ v8f wmma_f16(v16h a, v16h b, v8f c) {
  return __builtin_amdgcn_wmma_f32_16x16x32_f16(false, a, false, b,
                                                (short)0, c, false, false);
}

__device__ __forceinline__ v8h cvt8(v4f lo, v4f hi) {
  v8h h;
#pragma unroll
  for (int e = 0; e < 4; ++e) { h[e] = (_Float16)lo[e]; h[e + 4] = (_Float16)hi[e]; }
  return h;
}

// ---------------------------------------------------------------------------
// GEMM: C[M,N] = act( A[M,K] @ W + bias ), f16 WMMA, f32 accumulate.
//   WT=0: W stored [K,N] (reference [in,out]);  WT=1: W stored [N,K] (emb^T).
// Block tile 128x64, BK=32. 256 threads = 8 waves, each wave a 32x32 slab
// (2 A frags x 2 B frags = 4 WMMAs per K step). Software-pipelined staging.
// REQUIRES: M%128==0, N%64==0, K%32==0 (true for all call sites).
// ---------------------------------------------------------------------------
template <bool WT, bool RELU, bool HALF_OUT>
__global__ __launch_bounds__(256) void gemm_wmma_kernel(
    const float* __restrict__ A, const float* __restrict__ W,
    const float* __restrict__ bias, void* __restrict__ Cv,
    int M, int K, int N) {
  __shared__ _Float16 sA[128 * 32];   // [m][k]
  __shared__ _Float16 sB[64 * 32];    // [n][k]  (W transposed tile)
  const int tid = threadIdx.x, wave = tid >> 5, lane = tid & 31;
  const int bm = blockIdx.y * 128, bn = blockIdx.x * 64;
  const int wm = (wave >> 1) * 32, wn = (wave & 1) * 32;

  // A staging: 2 rows x 8 contiguous k per thread (global_load_b128 pairs)
  const int ar0 = tid >> 2, ar1 = ar0 + 64;
  const int ac  = (tid & 3) << 3;
  // W staging: WT=1 -> same pattern; WT=0 -> 8 contiguous n at one k (scatter)
  const int wr = WT ? (tid >> 2) : (tid & 31);
  const int wc = WT ? ((tid & 3) << 3) : ((tid >> 5) << 3);

  v4f a0l, a0h, a1l, a1h, w0l, w0h;
  const float* pa0 = A + (size_t)(bm + ar0) * K + ac;
  const float* pa1 = A + (size_t)(bm + ar1) * K + ac;
  const float* pw  = WT ? (W + (size_t)(bn + wr) * K + wc)
                        : (W + (size_t)wr * N + bn + wc);

  auto fetch = [&](int k0) {
    a0l = *(const v4f*)(pa0 + k0);     a0h = *(const v4f*)(pa0 + k0 + 4);
    a1l = *(const v4f*)(pa1 + k0);     a1h = *(const v4f*)(pa1 + k0 + 4);
    const float* p = WT ? (pw + k0) : (pw + (size_t)k0 * N);
    w0l = *(const v4f*)p;              w0h = *(const v4f*)(p + 4);
  };
  auto stash = [&]() {
    *(v8h*)(sA + ar0 * 32 + ac) = cvt8(a0l, a0h);
    *(v8h*)(sA + ar1 * 32 + ac) = cvt8(a1l, a1h);
    if (WT) {
      *(v8h*)(sB + wr * 32 + wc) = cvt8(w0l, w0h);
    } else {
#pragma unroll
      for (int e = 0; e < 4; ++e) {
        sB[(wc + e) * 32 + wr]     = (_Float16)w0l[e];
        sB[(wc + 4 + e) * 32 + wr] = (_Float16)w0h[e];
      }
    }
  };

  v8f acc00 = {}, acc01 = {}, acc10 = {}, acc11 = {};
  fetch(0);
  stash();
  __syncthreads();
  for (int k0 = 0; k0 < K; k0 += 32) {
    const bool more = (k0 + 32) < K;
    if (more) fetch(k0 + 32);          // overlap global latency with WMMAs
    v16h fa0 = load_frag(sA + wm * 32, 32);
    v16h fa1 = load_frag(sA + (wm + 16) * 32, 32);
    v16h fb0 = load_frag(sB + wn * 32, 32);
    v16h fb1 = load_frag(sB + (wn + 16) * 32, 32);
    acc00 = wmma_f16(fa0, fb0, acc00);
    acc01 = wmma_f16(fa0, fb1, acc01);
    acc10 = wmma_f16(fa1, fb0, acc10);
    acc11 = wmma_f16(fa1, fb1, acc11);
    __syncthreads();
    if (more) { stash(); __syncthreads(); }
  }

  const int mh = (lane >> 4) << 3, nc = lane & 15;
  const int n0 = bn + wn + nc, n1 = n0 + 16;
  const float bs0 = bias ? bias[n0] : 0.f;
  const float bs1 = bias ? bias[n1] : 0.f;
#pragma unroll
  for (int im = 0; im < 2; ++im) {
    const v8f& c0 = im ? acc10 : acc00;
    const v8f& c1 = im ? acc11 : acc01;
#pragma unroll
    for (int vv = 0; vv < 8; ++vv) {
      const int m = bm + wm + im * 16 + mh + vv;
      float r0 = c0[vv] + bs0, r1 = c1[vv] + bs1;
      if (RELU) { r0 = fmaxf(r0, 0.f); r1 = fmaxf(r1, 0.f); }
      if (HALF_OUT) {
        _Float16* C = (_Float16*)Cv;
        C[(size_t)m * N + n0] = (_Float16)r0;
        C[(size_t)m * N + n1] = (_Float16)r1;
      } else {
        float* C = (float*)Cv;
        C[(size_t)m * N + n0] = r0;
        C[(size_t)m * N + n1] = r1;
      }
    }
  }
}

// ---------------------------------------------------------------------------
// Flash attention: one wave per (batch, head, 16-query tile).
// q,k,v are f16 [B, S, H*HD]; o is f32 [B, Sq, H*HD].
// ---------------------------------------------------------------------------
template <bool CAUSAL>
__global__ __launch_bounds__(128) void attn_kernel(
    const _Float16* __restrict__ q, const _Float16* __restrict__ k,
    const _Float16* __restrict__ v, float* __restrict__ o,
    int Bn, int Hn, int Sq, int Sk, float scale) {
  __shared__ _Float16 sP[4][16 * 32];        // probs tile per wave
  __shared__ _Float16 sVt[4][HDIM * 32];     // V^T tile per wave: [hd][key]
  const int wave = threadIdx.x >> 5, lane = threadIdx.x & 31;
  const int nqt = Sq >> 4;
  const int task = blockIdx.x * 4 + wave;
  const int qt = task % nqt;
  const int h  = (task / nqt) % Hn;
  const int b  = task / (nqt * Hn);
  const int q0 = qt * 16;
  const int Dm = Hn * HDIM;
  const _Float16* qbase = q + ((size_t)(b * Sq + q0)) * Dm + h * HDIM;

  const int mh = (lane >> 4) << 3;
  const int nc = lane & 15;
  v8f oa[4] = {{}, {}, {}, {}};
  float mrow[8], lrow[8];
#pragma unroll
  for (int vv = 0; vv < 8; ++vv) { mrow[vv] = -1e30f; lrow[vv] = 0.f; }

  for (int kt = 0; kt < Sk; kt += 32) {
    if (CAUSAL && kt > q0 + 15) break;   // entire key tile masked
    const _Float16* kbase = k + ((size_t)(b * Sk + kt)) * Dm + h * HDIM;

    // S(16x32) = Q(16x64) @ K^T — K rows are contiguous => direct B^T frags
    v8f s0 = {}, s1 = {};
#pragma unroll
    for (int kc = 0; kc < HDIM; kc += 32) {
      v16h a  = load_frag(qbase + kc, Dm);
      v16h b0 = load_frag(kbase + kc, Dm);
      v16h b1 = load_frag(kbase + (size_t)16 * Dm + kc, Dm);
      s0 = wmma_f16(a, b0, s0);
      s1 = wmma_f16(a, b1, s1);
    }

    // stage V^T (32 keys x 64 hd -> [hd][key]) for the P@V WMMAs
#pragma unroll
    for (int it = 0; it < 8; ++it) {
      const int i = lane + it * 32;
      const int kl = i >> 3, hdc = (i & 7) << 3;
      v8h vv8 = *(const v8h*)(v + ((size_t)(b * Sk + kt + kl)) * Dm + h * HDIM + hdc);
#pragma unroll
      for (int e = 0; e < 8; ++e) sVt[wave][(hdc + e) * 32 + kl] = vv8[e];
    }

    // online softmax (row = mh+vv, spread over 16 lanes of this half-wave)
    float p0[8], p1[8], mcur[8];
#pragma unroll
    for (int vv = 0; vv < 8; ++vv) {
      float a0 = s0[vv] * scale, a1 = s1[vv] * scale;
      if (CAUSAL) {
        const int m = q0 + mh + vv;
        if (kt + nc > m)      a0 = -1e30f;
        if (kt + 16 + nc > m) a1 = -1e30f;
      }
      p0[vv] = a0; p1[vv] = a1;
      float mx = fmaxf(a0, a1);
#pragma unroll
      for (int msk = 1; msk < 16; msk <<= 1)
        mx = fmaxf(mx, __shfl_xor(mx, msk, 32));
      mcur[vv] = mx;
    }
#pragma unroll
    for (int vv = 0; vv < 8; ++vv) {
      const float mn = fmaxf(mrow[vv], mcur[vv]);
      const float alpha = __expf(mrow[vv] - mn);
      const float e0 = __expf(p0[vv] - mn);
      const float e1 = __expf(p1[vv] - mn);
      p0[vv] = e0; p1[vv] = e1;
      float s = e0 + e1;
#pragma unroll
      for (int msk = 1; msk < 16; msk <<= 1) s += __shfl_xor(s, msk, 32);
      lrow[vv] = lrow[vv] * alpha + s;
      mrow[vv] = mn;
#pragma unroll
      for (int j = 0; j < 4; ++j) oa[j][vv] *= alpha;
    }

    // P (C-layout) -> LDS -> A-layout fragment; O += P @ V
#pragma unroll
    for (int vv = 0; vv < 8; ++vv) {
      sP[wave][(mh + vv) * 32 + nc]      = (_Float16)p0[vv];
      sP[wave][(mh + vv) * 32 + nc + 16] = (_Float16)p1[vv];
    }
    asm volatile("s_wait_dscnt 0" ::: "memory");   // CDNA5 split DS counter
    v16h pa = load_frag(sP[wave], 32);
#pragma unroll
    for (int j = 0; j < 4; ++j) {
      v16h vb = load_frag(sVt[wave] + j * 16 * 32, 32);
      oa[j] = wmma_f16(pa, vb, oa[j]);
    }
  }

  float* obase = o + ((size_t)(b * Sq + q0)) * Dm + h * HDIM;
#pragma unroll
  for (int vv = 0; vv < 8; ++vv) {
    const float inv = 1.f / lrow[vv];
    const int m = mh + vv;
#pragma unroll
    for (int j = 0; j < 4; ++j)
      obase[(size_t)m * Dm + j * 16 + nc] = oa[j][vv] * inv;
  }
}

// ---------------------------------------------------------------------------
// out = LayerNorm(x + dlt) * g + b   (in-place safe when out == x)
// ---------------------------------------------------------------------------
__global__ __launch_bounds__(256) void add_ln_kernel(
    const float* __restrict__ x, const float* __restrict__ dlt,
    const float* __restrict__ g, const float* __restrict__ bta,
    float* __restrict__ out, int Dd) {
  __shared__ float red[256];
  const int row = blockIdx.x, tid = threadIdx.x;
  float acc = 0.f;
  for (int d = tid; d < Dd; d += 256)
    acc += x[(size_t)row * Dd + d] + dlt[(size_t)row * Dd + d];
  red[tid] = acc; __syncthreads();
  for (int s = 128; s > 0; s >>= 1) {
    if (tid < s) red[tid] += red[tid + s];
    __syncthreads();
  }
  const float mean = red[0] / (float)Dd;
  __syncthreads();
  float vac = 0.f;
  for (int d = tid; d < Dd; d += 256) {
    const float t = x[(size_t)row * Dd + d] + dlt[(size_t)row * Dd + d] - mean;
    vac += t * t;
  }
  red[tid] = vac; __syncthreads();
  for (int s = 128; s > 0; s >>= 1) {
    if (tid < s) red[tid] += red[tid + s];
    __syncthreads();
  }
  const float rstd = rsqrtf(red[0] / (float)Dd + 1e-5f);
  for (int d = tid; d < Dd; d += 256) {
    const float t = x[(size_t)row * Dd + d] + dlt[(size_t)row * Dd + d];
    out[(size_t)row * Dd + d] = (t - mean) * rstd * g[d] + bta[d];
  }
}

// ---------------------------------------------------------------------------
// out[row,d] = emb[tok[row], d] * sqrt(D) + sinusoidal_pos(row % S, d)
// ---------------------------------------------------------------------------
__global__ __launch_bounds__(256) void embed_kernel(
    const int* __restrict__ tok, const float* __restrict__ emb,
    float* __restrict__ out, int rows, int Slen, int Dd, float scale) {
  const int idx = blockIdx.x * blockDim.x + threadIdx.x;
  if (idx >= rows * Dd) return;
  const int row = idx / Dd, d = idx - row * Dd;
  const int s = row % Slen;
  const int j = d >> 1;
  const float omega = (float)(j + 1) / (float)Dd;
  const float wt = (float)(s + 1) * __expf(-omega * 9.210340371976184f); // ln(1e4)
  const float pe = (d & 1) ? __cosf(wt) : __sinf(wt);
  out[idx] = emb[(size_t)tok[row] * Dd + d] * scale + pe;
}

// ---------------------------------------------------------------------------
// Host orchestration
// ---------------------------------------------------------------------------
extern "C" void kernel_launch(void* const* d_in, const int* in_sizes, int n_in,
                              void* d_out, int out_size, void* d_ws, size_t ws_size,
                              hipStream_t stream) {
  (void)in_sizes; (void)n_in; (void)out_size; (void)ws_size;
  const int*   src      = (const int*)d_in[0];
  const int*   tgt      = (const int*)d_in[1];
  const float* src_emb  = (const float*)d_in[2];
  const float* tgt_emb  = (const float*)d_in[3];
  const float* enc_w    = (const float*)d_in[4];
  const float* enc_b    = (const float*)d_in[5];
  const float* enc_ff_w1 = (const float*)d_in[6];
  const float* enc_ff_b1 = (const float*)d_in[7];
  const float* enc_ff_w2 = (const float*)d_in[8];
  const float* enc_ff_b2 = (const float*)d_in[9];
  const float* enc_ln_g = (const float*)d_in[10];
  const float* enc_ln_b = (const float*)d_in[11];
  const float* dec_mw   = (const float*)d_in[12];
  const float* dec_mb   = (const float*)d_in[13];
  const float* dec_sw   = (const float*)d_in[14];
  const float* dec_sb   = (const float*)d_in[15];
  const float* dec_ff_w1 = (const float*)d_in[16];
  const float* dec_ff_b1 = (const float*)d_in[17];
  const float* dec_ff_w2 = (const float*)d_in[18];
  const float* dec_ff_b2 = (const float*)d_in[19];
  const float* dec_ln_g = (const float*)d_in[20];
  const float* dec_ln_b = (const float*)d_in[21];

  // workspace arena (all offsets 256B aligned)
  char* w = (char*)d_ws;
  float* xe = (float*)w;                 w += (size_t)MTOK * DM * 4;   // encoder activ
  float* xd = (float*)w;                 w += (size_t)MTOK * DM * 4;   // decoder activ
  float* ao = (float*)w;                 w += (size_t)MTOK * DM * 4;   // attn out
  float* t1 = (float*)w;                 w += (size_t)MTOK * FFD * 4;  // delta / ff hidden
  float* t2 = (float*)w;                 w += (size_t)MTOK * FFD * 4;
  _Float16* qh = (_Float16*)w;           w += (size_t)MTOK * DM * 2;
  _Float16* kh = (_Float16*)w;           w += (size_t)MTOK * DM * 2;
  _Float16* vh = (_Float16*)w;           w += (size_t)MTOK * DM * 2;

  const float embscale = 22.62741699796952f;   // sqrt(512)
  const float atscale = 0.125f;                // 1/sqrt(64)
  const dim3 blk256(256);
  const dim3 gEmb((MTOK * DM + 255) / 256);
  const dim3 gLN(MTOK);
  const dim3 gP(DM / 64, MTOK / 128);      // [1536,512] proj
  const dim3 gF1(FFD / 64, MTOK / 128);    // [1536,2048]
  const dim3 gF2(DM / 64, MTOK / 128);
  const dim3 gV(NVOCAB / 64, MTOK / 128);
  const int ntask = BBATCH * NH * (SSEQ / 16);
  const dim3 gAttn(ntask / 4), blkAttn(128);

  #define GEMM_QKV(A, W_, B_, C_) \
    gemm_wmma_kernel<false, false, true><<<gP, blk256, 0, stream>>>(A, W_, B_, (void*)(C_), MTOK, DM, DM)
  #define GEMM_O(A, W_, B_, C_) \
    gemm_wmma_kernel<false, false, false><<<gP, blk256, 0, stream>>>(A, W_, B_, (void*)(C_), MTOK, DM, DM)

  // ----- encoder -----
  embed_kernel<<<gEmb, blk256, 0, stream>>>(src, src_emb, xe, MTOK, SSEQ, DM, embscale);
  for (int i = 0; i < LYR; ++i) {
    const float* w4 = enc_w + (size_t)i * 4 * DM * DM;
    const float* b4 = enc_b + (size_t)i * 4 * DM;
    GEMM_QKV(xe, w4 + 0 * (size_t)DM * DM, b4 + 0 * DM, qh);
    GEMM_QKV(xe, w4 + 1 * (size_t)DM * DM, b4 + 1 * DM, kh);
    GEMM_QKV(xe, w4 + 2 * (size_t)DM * DM, b4 + 2 * DM, vh);
    attn_kernel<true><<<gAttn, blkAttn, 0, stream>>>(qh, kh, vh, ao, BBATCH, NH, SSEQ, SSEQ, atscale);
    GEMM_O(ao, w4 + 3 * (size_t)DM * DM, b4 + 3 * DM, t1);
    add_ln_kernel<<<gLN, blk256, 0, stream>>>(xe, t1, enc_ln_g, enc_ln_b, xe, DM);
    gemm_wmma_kernel<false, true, false><<<gF1, blk256, 0, stream>>>(
        xe, enc_ff_w1 + (size_t)i * DM * FFD, enc_ff_b1 + (size_t)i * FFD, (void*)t2, MTOK, DM, FFD);
    gemm_wmma_kernel<false, true, false><<<gF2, blk256, 0, stream>>>(
        t2, enc_ff_w2 + (size_t)i * FFD * DM, enc_ff_b2 + (size_t)i * DM, (void*)t1, MTOK, FFD, DM);
    add_ln_kernel<<<gLN, blk256, 0, stream>>>(xe, t1, enc_ln_g, enc_ln_b, xe, DM);
  }

  // ----- decoder -----
  embed_kernel<<<gEmb, blk256, 0, stream>>>(tgt, tgt_emb, xd, MTOK, TSEQ, DM, embscale);
  for (int i = 0; i < LYR; ++i) {
    const float* mw = dec_mw + (size_t)i * 4 * DM * DM;
    const float* mb = dec_mb + (size_t)i * 4 * DM;
    const float* sw = dec_sw + (size_t)i * 4 * DM * DM;
    const float* sb = dec_sb + (size_t)i * 4 * DM;
    // cross-attention (Q from decoder, K/V from encoder memory; non-causal)
    GEMM_QKV(xd, mw + 0 * (size_t)DM * DM, mb + 0 * DM, qh);
    GEMM_QKV(xe, mw + 1 * (size_t)DM * DM, mb + 1 * DM, kh);
    GEMM_QKV(xe, mw + 2 * (size_t)DM * DM, mb + 2 * DM, vh);
    attn_kernel<false><<<gAttn, blkAttn, 0, stream>>>(qh, kh, vh, ao, BBATCH, NH, TSEQ, SSEQ, atscale);
    GEMM_O(ao, mw + 3 * (size_t)DM * DM, mb + 3 * DM, t1);
    add_ln_kernel<<<gLN, blk256, 0, stream>>>(xd, t1, dec_ln_g, dec_ln_b, xd, DM);
    // causal self-attention
    GEMM_QKV(xd, sw + 0 * (size_t)DM * DM, sb + 0 * DM, qh);
    GEMM_QKV(xd, sw + 1 * (size_t)DM * DM, sb + 1 * DM, kh);
    GEMM_QKV(xd, sw + 2 * (size_t)DM * DM, sb + 2 * DM, vh);
    attn_kernel<true><<<gAttn, blkAttn, 0, stream>>>(qh, kh, vh, ao, BBATCH, NH, TSEQ, TSEQ, atscale);
    GEMM_O(ao, sw + 3 * (size_t)DM * DM, sb + 3 * DM, t1);
    add_ln_kernel<<<gLN, blk256, 0, stream>>>(xd, t1, dec_ln_g, dec_ln_b, xd, DM);
    // feed-forward (ReLU after BOTH linears, per reference)
    gemm_wmma_kernel<false, true, false><<<gF1, blk256, 0, stream>>>(
        xd, dec_ff_w1 + (size_t)i * DM * FFD, dec_ff_b1 + (size_t)i * FFD, (void*)t2, MTOK, DM, FFD);
    gemm_wmma_kernel<false, true, false><<<gF2, blk256, 0, stream>>>(
        t2, dec_ff_w2 + (size_t)i * FFD * DM, dec_ff_b2 + (size_t)i * DM, (void*)t1, MTOK, FFD, DM);
    add_ln_kernel<<<gLN, blk256, 0, stream>>>(xd, t1, dec_ln_g, dec_ln_b, xd, DM);
  }

  // ----- unembed: logits = y @ tgt_emb^T  (W stored [N,K] => WT=1) -----
  gemm_wmma_kernel<true, false, false><<<gV, blk256, 0, stream>>>(
      xd, tgt_emb, nullptr, d_out, MTOK, DM, NVOCAB);

  #undef GEMM_QKV
  #undef GEMM_O
}
